// PointNet_Graph_Vanilla_Parallel_81612968559258
// MI455X (gfx1250) — matrix operand order repaired and checked by reading, not compiled, not run on hardware
//
#include <hip/hip_runtime.h>

#define NP_  512
#define B_   8
#define N_   4096
#define K_   64
#define EPS_ 1e-5f

typedef _Float16 v16h __attribute__((ext_vector_type(16)));
typedef _Float16 v8h  __attribute__((ext_vector_type(8)));
typedef float    v8f  __attribute__((ext_vector_type(8)));

// ---------------------------------------------------------------------------
// KNN: one wave32 per dst point; candidates packed (d2bits<<32)|idx so integer
// min == (distance, index) lexicographic min; 64 rounds of butterfly reduction.
// ---------------------------------------------------------------------------
__global__ void knn_kernel(const float* __restrict__ pos, int* __restrict__ nbr)
{
    __shared__ float sp[NP_ * 3];
    const int tid = threadIdx.x;
    const int firstPoint = blockIdx.x * 8;      // 8 waves/block, never crosses a graph
    const int b = firstPoint / NP_;
    for (int t = tid; t < NP_ * 3; t += blockDim.x)
        sp[t] = pos[(size_t)b * NP_ * 3 + t];
    __syncthreads();

    const int wave = tid >> 5;
    const int lane = tid & 31;
    const int dst  = firstPoint + wave;         // global node id
    const int il   = dst - b * NP_;             // local point id

    const float px = sp[il * 3 + 0], py = sp[il * 3 + 1], pz = sp[il * 3 + 2];

    unsigned long long key[16];
#pragma unroll
    for (int s = 0; s < 16; ++s) {
        int c = lane + 32 * s;
        float dx = sp[c * 3 + 0] - px;
        float dy = sp[c * 3 + 1] - py;
        float dz = sp[c * 3 + 2] - pz;
        float d2 = dx * dx + dy * dy + dz * dz;
        key[s] = ((unsigned long long)__float_as_uint(d2) << 32) | (unsigned int)c;
    }

    for (int t = 0; t < K_; ++t) {
        unsigned long long m = key[0];
#pragma unroll
        for (int s = 1; s < 16; ++s) m = key[s] < m ? key[s] : m;
#pragma unroll
        for (int off = 16; off > 0; off >>= 1) {
            unsigned long long o = __shfl_xor(m, off, 32);
            m = o < m ? o : m;
        }
        int c = (int)(m & 0xffffffffu);
        if (lane == 0) nbr[(size_t)dst * K_ + t] = b * NP_ + c;
#pragma unroll
        for (int s = 0; s < 16; ++s)            // unrolled so key[] stays in VGPRs
            if (((c >> 5) == s) && ((c & 31) == lane))
                key[s] = ~0ULL;
    }
}

// ---------------------------------------------------------------------------
// GraphNorm (+ optional fused relu): one block per (graph, channel).
// ---------------------------------------------------------------------------
__global__ void gnorm_kernel(const float* __restrict__ x,
                             const float* __restrict__ gw,
                             const float* __restrict__ gb,
                             const float* __restrict__ gms,
                             float* __restrict__ y, int C, int doRelu)
{
    __shared__ float red[256];
    const int b = blockIdx.x, c = blockIdx.y, tid = threadIdx.x;
    const size_t base = (size_t)b * NP_;
    float v0 = x[(base + tid)       * C + c];
    float v1 = x[(base + tid + 256) * C + c];
    red[tid] = v0 + v1;
    __syncthreads();
    for (int s = 128; s > 0; s >>= 1) {
        if (tid < s) red[tid] += red[tid + s];
        __syncthreads();
    }
    float mean = red[0] * (1.0f / NP_);
    __syncthreads();
    float msc = gms[c];
    float o0 = v0 - mean * msc;
    float o1 = v1 - mean * msc;
    red[tid] = o0 * o0 + o1 * o1;
    __syncthreads();
    for (int s = 128; s > 0; s >>= 1) {
        if (tid < s) red[tid] += red[tid + s];
        __syncthreads();
    }
    float var   = red[0] * (1.0f / NP_);
    float scale = gw[c] * rsqrtf(var + EPS_);
    float bias  = gb[c];
    float r0 = scale * o0 + bias;
    float r1 = scale * o1 + bias;
    if (doRelu) { r0 = fmaxf(r0, 0.f); r1 = fmaxf(r1, 0.f); }
    y[(base + tid)       * C + c] = r0;
    y[(base + tid + 256) * C + c] = r1;
}

// ---------------------------------------------------------------------------
// Node-side GEMM:  a[n] = h[n]@w1[:Cin] + pos[n]@w1[Cin:] + b1 ;
//                  c[n] = pos[n]@w1[Cin:]          (both stored f16)
// ---------------------------------------------------------------------------
__global__ void node_gemm_kernel(const float* __restrict__ h,
                                 const float* __restrict__ pos,
                                 const float* __restrict__ w1,
                                 const float* __restrict__ b1,
                                 _Float16* __restrict__ a16,
                                 _Float16* __restrict__ c16,
                                 int Cin, int C)
{
    const int n  = blockIdx.x;
    const int co = blockIdx.y * 64 + threadIdx.x;
    float cc = 0.f;
#pragma unroll
    for (int d = 0; d < 3; ++d)
        cc += pos[(size_t)n * 3 + d] * w1[(size_t)(Cin + d) * C + co];
    float aa = b1[co] + cc;
    for (int k = 0; k < Cin; ++k)
        aa += h[(size_t)n * Cin + k] * w1[(size_t)k * C + co];
    a16[(size_t)n * C + co] = (_Float16)aa;
    c16[(size_t)n * C + co] = (_Float16)cc;
}

// w2 (C x C, f32, [k][n]) -> w2t (f16, [n][k]) so B fragments are contiguous.
__global__ void conv_w2_kernel(const float* __restrict__ w2,
                               _Float16* __restrict__ w2t, int C)
{
    int idx = blockIdx.x * blockDim.x + threadIdx.x;   // over C*C
    int n = idx / C, k = idx - n * C;
    w2t[idx] = (_Float16)w2[(size_t)k * C + n];
}

// ---------------------------------------------------------------------------
// Edge GEMM + fused scatter-max (the hot loop, ~320 GFLOP total):
// per dst i:  out[i] = colmax( relu(a[nbr(i)] - c[i]) @ w2 ) + b2
// Block = 8 waves, one dst per block, 16 output cols per wave.
// Two phases of 32 neighbor rows staged in LDS (f16, +8-half row pad).
// WMMA f32_16x16x32_f16, A and B both use contiguous K order k = 16*hi + s.
// ---------------------------------------------------------------------------
__global__ void edge_gemm_max_kernel(const _Float16* __restrict__ a16,
                                     const _Float16* __restrict__ c16,
                                     const _Float16* __restrict__ w2t,
                                     const float* __restrict__ b2,
                                     const int* __restrict__ nbr,
                                     float* __restrict__ out, int C)
{
    extern __shared__ _Float16 lds[];           // 32 x (C+8) halfs
    const int i      = blockIdx.x;
    const int tid    = threadIdx.x;
    const int stride = C + 8;
    const int wave   = tid >> 5;
    const int lane   = tid & 31;
    const int col    = blockIdx.y * 128 + wave * 16;
    const bool active = (col < C);              // wave-uniform

    const int n16 = lane & 15;
    const int hi  = lane >> 4;                  // 0 / 1
    const int kgA = hi * 16;                    // A K-group base
    const int kgB = hi * 16;                    // B K-group base

    const int* nb = nbr + (size_t)i * K_;
    const int  cw = C >> 3;                     // 8-half chunks per row
    const _Float16* bRow = w2t + (size_t)(col + n16) * C;

    v8f acc[4] = {};

#pragma unroll
    for (int ph = 0; ph < 2; ++ph) {
        __syncthreads();
        // stage rows [ph*32, ph*32+32): relu(a[j] - c[i]) in f16
        const int total = 32 * cw;
        for (int q = tid; q < total; q += blockDim.x) {
            int r  = q / cw;
            int kc = (q - r * cw) << 3;
            int j  = nb[ph * 32 + r];
            v8h av = *(const v8h*)(a16 + (size_t)j * C + kc);
            v8h cv = *(const v8h*)(c16 + (size_t)i * C + kc);
            v8h m;
#pragma unroll
            for (int e = 0; e < 8; ++e) {
                _Float16 t = av[e] - cv[e];
                m[e] = t > (_Float16)0.f ? t : (_Float16)0.f;
            }
            *(v8h*)(lds + (size_t)r * stride + kc) = m;
        }
        __syncthreads();

        if (active) {
            for (int k0 = 0; k0 < C; k0 += 32) {
                // B fragment: col = col+n16, K = k0+kgB .. +16 (contiguous in w2t)
                v8h b0 = *(const v8h*)(bRow + k0 + kgB);
                v8h b1 = *(const v8h*)(bRow + k0 + kgB + 8);
                v16h bf = __builtin_shufflevector(b0, b1,
                    0,1,2,3,4,5,6,7,8,9,10,11,12,13,14,15);
#pragma unroll
                for (int rt = 0; rt < 2; ++rt) {
                    // A fragment: M = rt*16+n16, K = kgA .. kgA+15 (contiguous)
                    const _Float16* ap = lds + (size_t)(rt * 16 + n16) * stride + k0 + kgA;
                    v8h a0 = *(const v8h*)ap;
                    v8h a1 = *(const v8h*)(ap + 8);
                    v16h af = __builtin_shufflevector(a0, a1,
                        0,1,2,3,4,5,6,7,8,9,10,11,12,13,14,15);
                    acc[ph * 2 + rt] = __builtin_amdgcn_wmma_f32_16x16x32_f16(
                        false, af, false, bf, (short)0, acc[ph * 2 + rt],
                        false, false);
                }
            }
        }
    }

    if (!active) return;

    // column max over all 64 neighbor rows:
    // per-lane max over 8 C/D VGPRs (8 M-rows), then xor-16 across M halves.
    float cmax = -3.402823466e38f;
#pragma unroll
    for (int rt = 0; rt < 4; ++rt)
#pragma unroll
        for (int r = 0; r < 8; ++r)
            cmax = fmaxf(cmax, acc[rt][r]);
    cmax = fmaxf(cmax, __shfl_xor(cmax, 16, 32));
    if (lane < 16)
        out[(size_t)i * C + col + n16] = cmax + b2[col + n16];
}

// ---------------------------------------------------------------------------
// Classifier: out = h @ cls_w + cls_b   (4096 x 512 x 14, trivial)
// ---------------------------------------------------------------------------
__global__ void cls_kernel(const float* __restrict__ h,
                           const float* __restrict__ w,
                           const float* __restrict__ bb,
                           float* __restrict__ out)
{
    int idx = blockIdx.x * blockDim.x + threadIdx.x;
    if (idx >= N_ * 14) return;
    int n = idx / 14, j = idx - n * 14;
    float s = bb[j];
    for (int k = 0; k < 512; ++k)
        s += h[(size_t)n * 512 + k] * w[k * 14 + j];
    out[idx] = s;
}

// ---------------------------------------------------------------------------
extern "C" void kernel_launch(void* const* d_in, const int* in_sizes, int n_in,
                              void* d_out, int out_size, void* d_ws, size_t ws_size,
                              hipStream_t stream)
{
    (void)in_sizes; (void)n_in; (void)out_size; (void)ws_size;

    const float* pos = (const float*)d_in[0];
    const float *gnw[6], *gnb[6], *gnms[6];
    for (int g = 0; g < 6; ++g) {
        gnw[g]  = (const float*)d_in[2 + 3 * g + 0];
        gnb[g]  = (const float*)d_in[2 + 3 * g + 1];
        gnms[g] = (const float*)d_in[2 + 3 * g + 2];
    }
    const float *w1s[5], *b1s[5], *w2s[5], *b2s[5];
    for (int l = 0; l < 5; ++l) {
        w1s[l] = (const float*)d_in[20 + 4 * l + 0];
        b1s[l] = (const float*)d_in[20 + 4 * l + 1];
        w2s[l] = (const float*)d_in[20 + 4 * l + 2];
        b2s[l] = (const float*)d_in[20 + 4 * l + 3];
    }
    const float* clsw = (const float*)d_in[40];
    const float* clsb = (const float*)d_in[41];
    float* out = (float*)d_out;

    char* ws = (char*)d_ws;
    int*      nbr = (int*)ws;                                  //  1 MB
    float*    hA  = (float*)(ws + (size_t)( 1 << 20));         //  8 MB
    float*    hB  = (float*)(ws + (size_t)( 9 << 20));         //  8 MB
    _Float16* a16 = (_Float16*)(ws + (size_t)(17 << 20));      //  4 MB
    _Float16* c16 = (_Float16*)(ws + (size_t)(21 << 20));      //  4 MB
    _Float16* w2t = (_Float16*)(ws + (size_t)(25 << 20));      // .5 MB

    knn_kernel<<<N_ / 8, 256, 0, stream>>>(pos, nbr);
    gnorm_kernel<<<dim3(B_, 3), 256, 0, stream>>>(pos, gnw[0], gnb[0], gnms[0], hA, 3, 0);

    const int Cs[5]   = {64, 128, 256, 512, 512};
    const int Cins[5] = {3, 64, 128, 256, 512};
    for (int l = 0; l < 5; ++l) {
        const int C = Cs[l], Cin = Cins[l];
        node_gemm_kernel<<<dim3(N_, C / 64), 64, 0, stream>>>(
            hA, pos, w1s[l], b1s[l], a16, c16, Cin, C);
        conv_w2_kernel<<<(C * C) / 256, 256, 0, stream>>>(w2s[l], w2t, C);
        size_t ldsB = (size_t)32 * (C + 8) * sizeof(_Float16);
        edge_gemm_max_kernel<<<dim3(N_, (C + 127) / 128), 256, ldsB, stream>>>(
            a16, c16, w2t, b2s[l], nbr, hB, C);
        gnorm_kernel<<<dim3(B_, C), 256, 0, stream>>>(
            hB, gnw[l + 1], gnb[l + 1], gnms[l + 1], hA, C, 1);
    }
    cls_kernel<<<(N_ * 14 + 255) / 256, 256, 0, stream>>>(hA, clsw, clsb, out);
}
